// GRU_Cell_16192026706568
// MI455X (gfx1250) — compile-verified, matrix operand order checked by reading, not run on hardware
//
#include <hip/hip_runtime.h>

// ---------------------------------------------------------------------------
// Problem constants: B=16, N=325, C=64, H=64, M=32, HID=32
//   positions P = B*N = 5200 = 325 tiles of 16
// ---------------------------------------------------------------------------
#define NTILES 325

typedef __attribute__((ext_vector_type(16))) __bf16   v16bf;
typedef __attribute__((ext_vector_type(8)))  float    v8f;
typedef __attribute__((ext_vector_type(4)))  unsigned int u32x4;

union FragU {
    u32x4        q[2];
    unsigned int u[8];
    v16bf        v;
};

__device__ __forceinline__ unsigned short f2bf(float f) {
    unsigned int u = __float_as_uint(f);
    unsigned int r = (u + 0x7FFFu + ((u >> 16) & 1u)) >> 16;
    return (unsigned short)r;
}

// ---------------------------------------------------------------------------
// Prelude: build bf16 WMMA B-operand fragments for the big GEMMs in d_ws.
//
// GEMM (per path in {x,h}, per gate g, per N-tile ht, per K-step kk):
//   B[k,n] (32x16) =
//     kk <  64 : W2[g][(kk*64 + ht*16 + n)*32 + k]          (weight term)
//     kk >= 64 : b2[g][ ((kk-64)*32 + k)*64 + ht*16 + n ]   (bias-matrix term)
//
// Per-lane element layout (16-bit WMMA operand, wave32):
//   lane l, element e:  col n = l&15,  K = (e<8 ? e : e+8) + (l<16 ? 0 : 8)
// Fragment f = (path*12 + g*4 + ht)*66 + kk ; 512 bf16 each (1KB).
// Total: 2*12*66 = 1584 fragments = 1,622,016 bytes in ws.
// ---------------------------------------------------------------------------
__global__ __launch_bounds__(256) void prep_frags(
    const float* __restrict__ wxW2, const float* __restrict__ wxb2,
    const float* __restrict__ whW2, const float* __restrict__ whb2,
    unsigned short* __restrict__ ws16)
{
    int tid = blockIdx.x * 256 + threadIdx.x;
    const int TOTAL = 1584 * 512;
    if (tid >= TOTAL) return;

    int e  = tid & 15;
    int l  = (tid >> 4) & 31;
    int f  = tid >> 9;          // fragment index
    int kk = f % 66;
    int q  = f / 66;            // path*12 + g*4 + ht
    int path = q / 12;
    int qq   = q % 12;
    int g    = qq >> 2;
    int ht   = qq & 3;

    int K = ((e < 8) ? e : e + 8) + ((l < 16) ? 0 : 8);
    int n = l & 15;

    const float* W2 = path ? whW2 : wxW2;
    const float* b2 = path ? whb2 : wxb2;

    float val;
    if (kk < 64)
        val = W2[(g * 4096 + kk * 64 + ht * 16 + n) * 32 + K];
    else
        val = b2[g * 4096 + ((kk - 64) * 32 + K) * 64 + ht * 16 + n];

    ws16[(size_t)f * 512 + l * 16 + e] = f2bf(val);
}

// ---------------------------------------------------------------------------
// One GEMM wave-task: acc(16x16) = sum_kk A(kk) * Bfrag(kk)
//   main loop (kk in [kc0,kc1), chunks of 4):
//       A = diag(scalar[:,kk]) * T   (8x v_pk_mul_bf16 on packed bf16)
//       row scalars fetched 4-at-a-time with one ds_load_b128
//   bias tail (kk = 64,65):  A = raw x/h tile slice (bias-matrix K-steps)
// ---------------------------------------------------------------------------
__device__ __forceinline__ void gemm_task(
    const unsigned short* __restrict__ wfrag,  // fragment base for (path,g,ht): 66 frags
    const unsigned short* __restrict__ Trow,   // T[g] tile: [16][32] bf16
    const unsigned int*   __restrict__ rep,    // scalar tile, bf16 replicated in u32: [16][64]
    const unsigned short* __restrict__ pl,     // plain bf16 tile: [16][64]
    float* __restrict__ dst,                   // output tile [16][64]
    int ht, int kc0, int kc1, bool do_bias, int lane)
{
    const int row  = lane & 15;
    const int half = lane >> 4;

    FragU tf;
    tf.q[0] = *(const u32x4*)(Trow + row * 32 + half * 8);
    tf.q[1] = *(const u32x4*)(Trow + row * 32 + half * 8 + 16);

    v8f acc;
    #pragma unroll
    for (int i = 0; i < 8; ++i) acc[i] = 0.0f;

    const unsigned short* wp = wfrag + (size_t)kc0 * 512 + lane * 16;
    for (int kc = kc0; kc < kc1; kc += 4) {
        u32x4 s4 = *(const u32x4*)(rep + row * 64 + kc);
        #pragma unroll
        for (int j = 0; j < 4; ++j) {
            FragU a, b;
            const u32x4* bp = (const u32x4*)wp;
            b.q[0] = bp[0];
            b.q[1] = bp[1];
            unsigned int s = s4[j];
            #pragma unroll
            for (int i = 0; i < 8; ++i)
                asm("v_pk_mul_bf16 %0, %1, %2" : "=v"(a.u[i]) : "v"(tf.u[i]), "v"(s));
            acc = __builtin_amdgcn_wmma_f32_16x16x32_bf16(
                      false, a.v, false, b.v, (short)0, acc, false, false);
            wp += 512;
        }
    }

    if (do_bias) {
        #pragma unroll
        for (int t = 0; t < 2; ++t) {
            FragU a, b;
            const u32x4* bp = (const u32x4*)(wfrag + (size_t)(64 + t) * 512 + lane * 16);
            b.q[0] = bp[0];
            b.q[1] = bp[1];
            const unsigned short* xp = pl + row * 64 + t * 32 + half * 8;
            a.q[0] = *(const u32x4*)xp;
            a.q[1] = *(const u32x4*)(xp + 16);
            acc = __builtin_amdgcn_wmma_f32_16x16x32_bf16(
                      false, a.v, false, b.v, (short)0, acc, false, false);
        }
    }

    // C/D layout: lane holds column n=row; rows M = half*8 + v
    const int n = ht * 16 + row;
    #pragma unroll
    for (int v = 0; v < 8; ++v)
        dst[(half * 8 + v) * 64 + n] = acc[v];
}

// ---------------------------------------------------------------------------
// Main kernel: one workgroup (8 waves) per tile of 16 positions.
// ---------------------------------------------------------------------------
__global__ __launch_bounds__(256) void gru_tile(
    const float* __restrict__ x,     const float* __restrict__ state,
    const float* __restrict__ meta,
    const float* __restrict__ wxW1,  const float* __restrict__ wxb1,
    const float* __restrict__ whW1,  const float* __restrict__ whb1,
    const float* __restrict__ bW1,   const float* __restrict__ bb1,
    const float* __restrict__ bW2,   const float* __restrict__ bb2,
    const unsigned short* __restrict__ ws16,
    float* __restrict__ out)
{
    __shared__ __align__(16) float          meta_s[16][32];
    __shared__ __align__(16) float          Tb_s[3][16][32];
    __shared__ __align__(16) unsigned short Tx_s[3][16][32];
    __shared__ __align__(16) unsigned short Th_s[3][16][32];
    __shared__ __align__(16) unsigned int   xrep_s[16][64];   // reused for r*h in phase C
    __shared__ __align__(16) unsigned int   hrep_s[16][64];
    __shared__ __align__(16) unsigned short xbf_s[16][64];    // reused for r*h in phase C
    __shared__ __align__(16) unsigned short hbf_s[16][64];
    __shared__ __align__(16) float          xW_s[3][16][64];
    __shared__ __align__(16) float          hW_s[2][16][64];
    __shared__ __align__(16) float          bbv_s[3][16][64];
    __shared__ __align__(16) float          hcp_s[2][16][64];
    __shared__ __align__(16) float          h_s[16][64];
    __shared__ __align__(16) float          z_s[16][64];

    const int tid  = threadIdx.x;
    const int wave = tid >> 5;
    const int lane = tid & 31;
    const int base = blockIdx.x * 1024;   // 16 positions * 64
    const int mbase = blockIdx.x * 512;   // 16 positions * 32

    // ---------------- Phase 0: load tile, convert to bf16 forms ------------
    for (int idx = tid; idx < 512; idx += 256)
        meta_s[idx >> 5][idx & 31] = meta[mbase + idx];
    for (int idx = tid; idx < 1024; idx += 256) {
        int p = idx >> 6, c = idx & 63;
        float xv = x[base + idx];
        unsigned short ux = f2bf(xv);
        xrep_s[p][c] = (unsigned int)ux * 0x10001u;
        xbf_s[p][c]  = ux;
        float hv = state[base + idx];
        unsigned short uh = f2bf(hv);
        h_s[p][c]    = hv;
        hrep_s[p][c] = (unsigned int)uh * 0x10001u;
        hbf_s[p][c]  = uh;
    }
    __syncthreads();

    // ---------------- Phase 1: hyper layer-1, T = relu(meta@W1^T + b1) -----
    for (int idx = tid; idx < 4608; idx += 256) {
        int k = idx & 31, p = (idx >> 5) & 15, r = idx >> 9;
        int g = r % 3, net = r / 3;
        const float* W1 = (net == 0) ? wxW1 : (net == 1) ? whW1 : bW1;
        const float* B1 = (net == 0) ? wxb1 : (net == 1) ? whb1 : bb1;
        float acc = B1[g * 32 + k];
        const float* wrow = &W1[(g * 32 + k) * 32];
        #pragma unroll
        for (int m = 0; m < 32; ++m) acc += meta_s[p][m] * wrow[m];
        acc = fmaxf(acc, 0.0f);
        if (net == 0)      Tx_s[g][p][k] = f2bf(acc);
        else if (net == 1) Th_s[g][p][k] = f2bf(acc);
        else               Tb_s[g][p][k] = acc;
    }
    __syncthreads();

    // ---------------- Phase 1b: bias net output bb = Tb@bW2^T + bb2 --------
    for (int idx = tid; idx < 3072; idx += 256) {
        int hh = idx & 63, p = (idx >> 6) & 15, g = idx >> 10;
        float acc = bb2[g * 64 + hh];
        const float* w = &bW2[(g * 64 + hh) * 32];
        #pragma unroll
        for (int k = 0; k < 32; ++k) acc += Tb_s[g][p][k] * w[k];
        bbv_s[g][p][hh] = acc;
    }

    // ---------------- Phase 2: WMMA GEMMs (x-path g=0..2, h-path g=0,1) ----
    for (int task = wave; task < 20; task += 8) {
        int path  = (task < 12) ? 0 : 1;
        int local = (task < 12) ? task : task - 12;
        int g  = local >> 2;
        int ht = local & 3;
        const unsigned short* wfrag =
            ws16 + (size_t)((path * 12 + g * 4 + ht) * 66) * 512;
        gemm_task(wfrag,
                  path ? &Th_s[g][0][0] : &Tx_s[g][0][0],
                  path ? &hrep_s[0][0]  : &xrep_s[0][0],
                  path ? &hbf_s[0][0]   : &xbf_s[0][0],
                  path ? &hW_s[g][0][0] : &xW_s[g][0][0],
                  ht, 0, 64, true, lane);
    }
    __syncthreads();

    // ---------------- Phase RZ: gates r,z; stash r*h for candidate GEMM ----
    for (int idx = tid; idx < 1024; idx += 256) {
        int p = idx >> 6, hh = idx & 63;
        float hv    = h_s[p][hh];
        float pre_r = xW_s[0][p][hh] + hW_s[0][p][hh] + bbv_s[0][p][hh];
        float pre_z = xW_s[1][p][hh] + hW_s[1][p][hh] + bbv_s[1][p][hh];
        float rg = 1.0f / (1.0f + __expf(-pre_r));
        float zg = 1.0f / (1.0f + __expf(-pre_z));
        z_s[p][hh] = zg;
        float rh = rg * hv;
        unsigned short us = f2bf(rh);
        xrep_s[p][hh] = (unsigned int)us * 0x10001u;   // reuse as rh_rep
        xbf_s[p][hh]  = us;                            // reuse as rh_bf
    }
    __syncthreads();

    // ---------------- Phase C: candidate GEMM (r*h)@Wh[2], K split 2-way ---
    {
        int ht = wave & 3, kh = wave >> 2;
        const unsigned short* wfrag =
            ws16 + (size_t)((1 * 12 + 2 * 4 + ht) * 66) * 512;
        gemm_task(wfrag, &Th_s[2][0][0], &xrep_s[0][0], &xbf_s[0][0],
                  &hcp_s[kh][0][0], ht,
                  kh ? 32 : 0, kh ? 64 : 32, kh != 0, lane);
    }
    __syncthreads();

    // ---------------- Phase D: hc = tanh(...), h_new = (1-z)h + z*hc -------
    for (int idx = tid; idx < 1024; idx += 256) {
        int p = idx >> 6, hh = idx & 63;
        float pre = xW_s[2][p][hh] + hcp_s[0][p][hh] + hcp_s[1][p][hh]
                  + bbv_s[2][p][hh];
        float ax = fabsf(pre);
        float e2 = __expf(-2.0f * ax);
        float th = (1.0f - e2) / (1.0f + e2);
        th = (pre < 0.0f) ? -th : th;
        float zg = z_s[p][hh];
        float hv = h_s[p][hh];
        out[base + idx] = (1.0f - zg) * hv + zg * th;
    }
}

// ---------------------------------------------------------------------------
extern "C" void kernel_launch(void* const* d_in, const int* in_sizes, int n_in,
                              void* d_out, int out_size, void* d_ws, size_t ws_size,
                              hipStream_t stream) {
    const float* x     = (const float*)d_in[0];
    const float* state = (const float*)d_in[1];
    const float* meta  = (const float*)d_in[2];
    const float* wxW1  = (const float*)d_in[3];
    const float* wxb1  = (const float*)d_in[4];
    const float* wxW2  = (const float*)d_in[5];
    const float* wxb2  = (const float*)d_in[6];
    const float* whW1  = (const float*)d_in[7];
    const float* whb1  = (const float*)d_in[8];
    const float* whW2  = (const float*)d_in[9];
    const float* whb2  = (const float*)d_in[10];
    const float* bW1   = (const float*)d_in[11];
    const float* bb1   = (const float*)d_in[12];
    const float* bW2   = (const float*)d_in[13];
    const float* bb2   = (const float*)d_in[14];
    unsigned short* ws16 = (unsigned short*)d_ws;
    float* out = (float*)d_out;

    const int prep_total = 1584 * 512;
    prep_frags<<<(prep_total + 255) / 256, 256, 0, stream>>>(
        wxW2, wxb2, whW2, whb2, ws16);
    gru_tile<<<NTILES, 256, 0, stream>>>(
        x, state, meta, wxW1, wxb1, whW1, whb1, bW1, bb1, bW2, bb2, ws16, out);
}
